// TransformerPolicyNetwork_48369921688020
// MI455X (gfx1250) — matrix operand order repaired and checked by reading, not compiled, not run on hardware
//
#include <hip/hip_runtime.h>
#include <math.h>

// ---------------------------------------------------------------------------
// CDNA5 (gfx1250) wave32 WMMA implementation of TransformerPolicyNetwork.
// Heavy matmuls: v_wmma_f32_16x16x32_f16 (f32 accumulate).
// Data movement: global_load_async_to_lds_b128 (ASYNCcnt) for f16 activation
// tiles, TDM tensor_load_to_lds (TENSORcnt) for attention K tiles.
// ---------------------------------------------------------------------------

typedef __attribute__((ext_vector_type(16))) _Float16 v16h;
typedef __attribute__((ext_vector_type(8)))  float    v8f;
typedef __attribute__((ext_vector_type(4)))  unsigned u32x4;
typedef __attribute__((ext_vector_type(8)))  unsigned u32x8;

__device__ __forceinline__ v8f wmma_f16(v16h a, v16h b, v8f c) {
  return __builtin_amdgcn_wmma_f32_16x16x32_f16(false, a, false, b, (short)0, c,
                                                false, false);
}

// Async global->LDS 16-byte copy (per-lane). LDS offset = low 32 bits of the
// generic pointer (ISA 10.2).
__device__ __forceinline__ void async_g2l_b128(const void* gp, void* lp) {
  unsigned       ldsa = (unsigned)(unsigned long long)lp;
  unsigned long long ga = (unsigned long long)gp;
  asm volatile("global_load_async_to_lds_b128 %0, %1, off"
               :: "v"(ldsa), "v"(ga) : "memory");
}
__device__ __forceinline__ void wait_asynccnt0() {
  asm volatile("s_wait_asynccnt 0x0" ::: "memory");
}
__device__ __forceinline__ void wait_tensorcnt0() {
  asm volatile("s_wait_tensorcnt 0x0" ::: "memory");
}
__device__ __forceinline__ void wait_dscnt0() {
  asm volatile("s_wait_dscnt 0x0" ::: "memory");
}

// ---------------------------------------------------------------------------
// Tiled GEMM:  C[M,N] = A[M,K] (f32 or f16) * B[K,N] (f32 weights) + bias
// Block tile 128x128, K tile 32. 8 wave32 waves as 4(M) x 2(N); each wave
// computes 32x64 via eight 16x16x32 WMMAs per K step (B frags hoisted).
// A tile staged via async global->LDS when A is f16 and tiles are full.
// ---------------------------------------------------------------------------
template <typename AT>
__global__ __launch_bounds__(256) void gemm_wmma(
    const AT* __restrict__ A, int lda, const float* __restrict__ B, int ldb,
    const float* __restrict__ bias, float* __restrict__ Cf,
    _Float16* __restrict__ Ch, int ldc, int M, int N, int K) {
  constexpr int BM = 128, BN = 128, BK = 32;
  // strides padded to 48 halfs (96B: 16B-aligned rows for b128 LDS traffic)
  __shared__ _Float16 As[BM][BK + 16];
  __shared__ _Float16 Bt[BN][BK + 16];   // transposed: Bt[n][k]

  const int tid  = threadIdx.x;
  const int lane = tid & 31;
  const int wave = tid >> 5;  // 0..7
  const int wm   = wave >> 1; // 0..3 -> 32-row subtile
  const int wn   = wave & 1;  // 0..1 -> 64-col subtile
  const int m0   = blockIdx.y * BM;
  const int n0   = blockIdx.x * BN;
  const int g    = lane >> 4; // lane half
  const int lm   = lane & 15;

  v8f acc[2][4];
#pragma unroll
  for (int i = 0; i < 2; ++i)
#pragma unroll
    for (int j = 0; j < 4; ++j)
#pragma unroll
      for (int r = 0; r < 8; ++r) acc[i][j][r] = 0.0f;

  const bool fullA = ((M & (BM - 1)) == 0) && ((K & (BK - 1)) == 0);

  for (int k0 = 0; k0 < K; k0 += BK) {
    // ---- stage A tile 128x32 ----
    if (sizeof(AT) == 2 && fullA) {
      // async DMA path: 512 x b128 chunks, 2 per thread
#pragma unroll
      for (int i = 0; i < 2; ++i) {
        int c    = i * 256 + tid;
        int row  = c >> 2;
        int col8 = (c & 3) * 8;
        async_g2l_b128(A + (size_t)(m0 + row) * lda + k0 + col8,
                       &As[row][col8]);
      }
    } else {
      int r  = tid >> 1;
      int cb = (tid & 1) * 16;
      int gr = m0 + r;
#pragma unroll
      for (int c = 0; c < 16; ++c) {
        int   gk = k0 + cb + c;
        float v  = 0.0f;
        if (gr < M && gk < K) v = (float)A[(size_t)gr * lda + gk];
        As[r][cb + c] = (_Float16)v;
      }
    }
    // ---- stage B tile 32x128 transposed (f32 -> f16) ----
    {
      int r  = tid >> 3;        // k row 0..31
      int cb = (tid & 7) * 16;  // n 0..127
      int gk = k0 + r;
#pragma unroll
      for (int c = 0; c < 16; ++c) {
        int   gn = n0 + cb + c;
        float v  = 0.0f;
        if (gk < K && gn < N) v = B[(size_t)gk * ldb + gn];
        Bt[cb + c][r] = (_Float16)v;
      }
    }
    // CDNA5 speculative prefetch of next weight tile -> global_prefetch_b8
    if (tid < 32 && (k0 + BK + tid) < K)
      __builtin_prefetch(&B[(size_t)(k0 + BK + tid) * ldb + n0], 0, 1);
    if (sizeof(AT) == 2 && fullA) wait_asynccnt0();
    __syncthreads();

    // hoisted B fragments (table k order, matching A)
    union bfu { v16h v; unsigned u[8]; } bfr[4];
#pragma unroll
    for (int j = 0; j < 4; ++j) {
      const _Float16* bp = &Bt[wn * 64 + j * 16 + lm][g * 8];
#pragma unroll
      for (int p = 0; p < 8; ++p) bfr[j].u[p] = *(const unsigned*)(bp + (p >> 2) * 16 + (p & 3) * 2);
    }
#pragma unroll
    for (int i = 0; i < 2; ++i) {
      union { v16h v; unsigned u[8]; } af;  // ISA 16-bit 16x32 A layout
      int ar = wm * 32 + i * 16 + lm;
#pragma unroll
      for (int p = 0; p < 8; ++p) {
        int kk  = (p >> 2) * 16 + g * 8 + (p & 3) * 2;
        af.u[p] = *(const unsigned*)&As[ar][kk];
      }
#pragma unroll
      for (int j = 0; j < 4; ++j) acc[i][j] = wmma_f16(af.v, bfr[j].v, acc[i][j]);
    }
    __syncthreads();
  }

  // ---- epilogue: bias + write f32 and/or f16 (C layout: row=r+8*half) ----
#pragma unroll
  for (int i = 0; i < 2; ++i)
#pragma unroll
    for (int j = 0; j < 4; ++j) {
      int rowb = m0 + wm * 32 + i * 16 + g * 8;
      int col  = n0 + wn * 64 + j * 16 + lm;
      if (col < N) {
        float bv = bias ? bias[col] : 0.0f;
#pragma unroll
        for (int r = 0; r < 8; ++r) {
          int row = rowb + r;
          if (row < M) {
            float v = acc[i][j][r] + bv;
            if (Cf) Cf[(size_t)row * ldc + col] = v;
            if (Ch) Ch[(size_t)row * ldc + col] = (_Float16)v;
          }
        }
      }
    }
}

// ---------------------------------------------------------------------------
// Flash attention (causal, per-head temp bias).  One block = (b,h, 64 q rows),
// 4 wave32 waves, 16 q rows each.  dk = 64.  K tiles of 32.
// K tile staged by TDM (tensor_load_to_lds); V tile staged transposed.
// ---------------------------------------------------------------------------
__global__ __launch_bounds__(128) void attn_wmma(
    const _Float16* __restrict__ Q, const _Float16* __restrict__ Kb,
    const _Float16* __restrict__ Vb, _Float16* __restrict__ Ctx,
    const float* __restrict__ temp_bias, int S, int D, int H) {
  constexpr int AQT = 64, AKT = 32;
  const int dk   = D / H;           // 64
  const int bh   = blockIdx.x;
  const int b    = bh / H;
  const int h    = bh % H;
  const int q0   = blockIdx.y * AQT;
  const int tid  = threadIdx.x;
  const int lane = tid & 31;
  const int wave = tid >> 5;        // 0..3
  const int g    = lane >> 4;
  const int lm   = lane & 15;

  __shared__ _Float16 Ks[AKT][64 + 8];     // row stride 72 halfs = 36 DWORDs
  __shared__ _Float16 Vt[64][AKT + 16];    // transposed V: Vt[d][k]
  __shared__ _Float16 Ps[4][16][AKT + 16]; // per-wave P re-layout buffer

  const size_t baseQ = ((size_t)b * S) * D + (size_t)h * dk;

  // Constant part of TDM descriptor (D#), ISA 8.3-8.6:
  // 2D tile: tile_dim0=64, tile_dim1=32, tensor_dim0=64, tensor_dim1=32,
  // dim0 stride = D elems, data_size=2B, LDS pad 4 DWORDs per 32 DWORDs.
  u32x8 g1;
  g1[0] = (1u << 16)      // data_size = 1 -> 2 bytes
        | (1u << 20)      // pad_enable
        | (4u << 22)      // pad_interval: 32 DWORDs
        | (3u << 25);     // pad_amount:   4 DWORDs
  g1[1] = (64u & 0xffffu) << 16;        // tensor_dim0 = 64 (low 16 in [63:48])
  g1[2] = (32u & 0xffffu) << 16;        // tensor_dim1 = 32 (low 16 in [95:80])
  g1[3] = 64u << 16;                    // tile_dim0 = 64  ([127:112])
  g1[4] = 32u;                          // tile_dim1 = 32  ([143:128])
  g1[5] = (unsigned)D;                  // tensor_dim0_stride low 32
  g1[6] = 0u;
  g1[7] = 0u;
  u32x4 gz; gz[0] = 0; gz[1] = 0; gz[2] = 0; gz[3] = 0;
  const unsigned ksLds = (unsigned)(unsigned long long)&Ks[0][0];

  // preload Q fragments (dk chunks 0..31 and 32..63) in A-matrix layout
  union { v16h v; unsigned u[8]; } qf[2];
  {
    int qr = q0 + wave * 16 + lm;
    const _Float16* qp = Q + baseQ + (size_t)qr * D;
#pragma unroll
    for (int f = 0; f < 2; ++f)
#pragma unroll
      for (int p = 0; p < 8; ++p) {
        int kk   = f * 32 + (p >> 2) * 16 + g * 8 + (p & 3) * 2;
        qf[f].u[p] = *(const unsigned*)(qp + kk);
      }
  }

  float mrun[8], lrun[8];
  v8f   o[4];
#pragma unroll
  for (int r = 0; r < 8; ++r) { mrun[r] = -1e30f; lrun[r] = 0.0f; }
#pragma unroll
  for (int j = 0; j < 4; ++j)
#pragma unroll
    for (int r = 0; r < 8; ++r) o[j][r] = 0.0f;

  const float scale = 0.125f;       // 1/sqrt(64)
  const float tb    = temp_bias[h];

  for (int k0 = 0; k0 < q0 + AQT; k0 += AKT) {
    // ---- stage K tile via Tensor Data Mover (issued once, by wave 0) ----
    if (wave == 0) {
      unsigned long long ga =
          (unsigned long long)(Kb + baseQ + (size_t)k0 * D);
      u32x4 g0;
      g0[0] = 1u;                                   // count = 1 (valid)
      g0[1] = ksLds;                                // lds_addr
      g0[2] = (unsigned)ga;                         // global_addr[31:0]
      g0[3] = (unsigned)((ga >> 32) & 0x01ffffffu)  // global_addr[56:32]
            | (2u << 30);                           // type = 2 (image)
      asm volatile("tensor_load_to_lds %0, %1, %2, %3"
                   :: "s"(g0), "s"(g1), "s"(gz), "s"(gz) : "memory");
      wait_tensorcnt0();
    }
    // ---- stage V tile transposed (manual) ----
    {
      int r  = tid >> 2;        // k row 0..31
      int cb = (tid & 3) * 16;  // d 0..63
      const _Float16* vp = Vb + baseQ + (size_t)(k0 + r) * D + cb;
#pragma unroll
      for (int c = 0; c < 16; ++c) Vt[cb + c][r] = vp[c];
    }
    __syncthreads();

    // ---- scores: two 16x16 tiles, contraction over dk=64 (2 chained WMMAs) ----
    v8f s[2];
#pragma unroll
    for (int t = 0; t < 2; ++t) {
#pragma unroll
      for (int r = 0; r < 8; ++r) s[t][r] = 0.0f;
#pragma unroll
      for (int f = 0; f < 2; ++f) {
        union { v16h v; unsigned u[8]; } kf; // B = K^T: lane = k col, elems = dk
        int kcol = t * 16 + lm;
#pragma unroll
        for (int p = 0; p < 8; ++p) {
          int dd  = f * 32 + (p >> 2) * 16 + g * 8 + (p & 3) * 2;
          kf.u[p] = *(const unsigned*)&Ks[kcol][dd];
        }
        s[t] = wmma_f16(qf[f].v, kf.v, s[t]);
      }
    }

    // ---- scale + temp bias + causal mask ----
    const int rowbase = q0 + wave * 16 + 8 * g;
#pragma unroll
    for (int t = 0; t < 2; ++t)
#pragma unroll
      for (int r = 0; r < 8; ++r) {
        int   row = rowbase + r;
        int   col = k0 + t * 16 + lm;
        float v   = s[t][r] * scale + tb;
        if (col > row) v = -1e30f;
        s[t][r] = v;
      }

    // ---- online softmax (row reductions across 16-lane halves) ----
#pragma unroll
    for (int r = 0; r < 8; ++r) {
      float v = fmaxf(s[0][r], s[1][r]);
#pragma unroll
      for (int off = 8; off >= 1; off >>= 1) v = fmaxf(v, __shfl_xor(v, off, 32));
      float mn    = fmaxf(mrun[r], v);
      float alpha = __expf(mrun[r] - mn);
      mrun[r]     = mn;
      float psum  = 0.0f;
#pragma unroll
      for (int t = 0; t < 2; ++t) {
        float p  = __expf(s[t][r] - mn);
        s[t][r]  = p;
        psum    += p;
      }
#pragma unroll
      for (int off = 8; off >= 1; off >>= 1) psum += __shfl_xor(psum, off, 32);
      lrun[r] = lrun[r] * alpha + psum;
#pragma unroll
      for (int j = 0; j < 4; ++j) o[j][r] *= alpha;
    }

    // ---- re-layout P (C layout -> A layout) through per-wave LDS ----
#pragma unroll
    for (int t = 0; t < 2; ++t)
#pragma unroll
      for (int r = 0; r < 8; ++r)
        Ps[wave][r + 8 * g][t * 16 + lm] = (_Float16)s[t][r];
    wait_dscnt0(); // same-wave LDS RAW

    union { v16h v; unsigned u[8]; } pf;
#pragma unroll
    for (int p = 0; p < 8; ++p) {
      int kk  = (p >> 2) * 16 + g * 8 + (p & 3) * 2;
      pf.u[p] = *(const unsigned*)&Ps[wave][lm][kk];
    }

    // ---- O += P @ V  (4 output column tiles of 16, Vt reads contiguous) ----
#pragma unroll
    for (int j = 0; j < 4; ++j) {
      union { v16h v; unsigned u[8]; } vf;
      const _Float16* vtp = &Vt[j * 16 + lm][g * 8];
#pragma unroll
      for (int p = 0; p < 8; ++p) vf.u[p] = *(const unsigned*)(vtp + (p >> 2) * 16 + (p & 3) * 2);
      o[j] = wmma_f16(pf.v, vf.v, o[j]);
    }
    __syncthreads();
  }

  // ---- normalize and write ctx (f16) ----
#pragma unroll
  for (int j = 0; j < 4; ++j)
#pragma unroll
    for (int r = 0; r < 8; ++r) {
      int   row = q0 + wave * 16 + r + 8 * g;
      float inv = 1.0f / lrun[r];
      Ctx[((size_t)b * S + row) * D + h * dk + j * 16 + lm] =
          (_Float16)(o[j][r] * inv);
    }
}

// ---------------------------------------------------------------------------
// emb + sinusoidal PE + learnable PE  ->  hid (f32) and hid_h (f16)
// ---------------------------------------------------------------------------
__global__ void pe_add(const float* __restrict__ emb, const float* __restrict__ lpe,
                       float* __restrict__ hid, _Float16* __restrict__ hidh,
                       int S, int D, int total) {
  int idx = blockIdx.x * blockDim.x + threadIdx.x;
  if (idx >= total) return;
  int d = idx % D;
  int s = (idx / D) % S;
  int i = d >> 1;
  float div = __expf(-(float)(2 * i) * (logf(10000.0f) / (float)D));
  float arg = (float)s * div;
  float pe  = (d & 1) ? __cosf(arg) : __sinf(arg);
  float v   = emb[idx] + pe + lpe[(size_t)s * D + d];
  hid[idx]  = v;
  hidh[idx] = (_Float16)v;
}

// ---------------------------------------------------------------------------
// hid = LN(X + Res) * g + b ; also writes f16 mirror.  One row per block(128).
// ---------------------------------------------------------------------------
__global__ __launch_bounds__(128) void add_ln(
    const float* __restrict__ X, const float* __restrict__ Res,
    const float* __restrict__ gam, const float* __restrict__ bet,
    float* __restrict__ hid, _Float16* __restrict__ hidh, int D) {
  const int row = blockIdx.x;
  const int tid = threadIdx.x;
  const float* xr = X + (size_t)row * D;
  const float* rr = Res + (size_t)row * D;
  float vals[4], s1 = 0.0f, s2 = 0.0f;
#pragma unroll
  for (int i = 0; i < 4; ++i) {
    float v = xr[tid + i * 128] + rr[tid + i * 128];
    vals[i] = v; s1 += v; s2 += v * v;
  }
#pragma unroll
  for (int off = 16; off >= 1; off >>= 1) {
    s1 += __shfl_xor(s1, off, 32);
    s2 += __shfl_xor(s2, off, 32);
  }
  __shared__ float a1[4], a2[4];
  int w = tid >> 5, ln = tid & 31;
  if (ln == 0) { a1[w] = s1; a2[w] = s2; }
  __syncthreads();
  s1 = a1[0] + a1[1] + a1[2] + a1[3];
  s2 = a2[0] + a2[1] + a2[2] + a2[3];
  float mean = s1 / (float)D;
  float var  = s2 / (float)D - mean * mean;
  float rstd = rsqrtf(var + 1e-5f);
#pragma unroll
  for (int i = 0; i < 4; ++i) {
    int   c = tid + i * 128;
    float v = (vals[i] - mean) * rstd * gam[c] + bet[c];
    hid[(size_t)row * D + c]  = v;
    hidh[(size_t)row * D + c] = (_Float16)v;
  }
}

// g = gelu(ff) * sigmoid(gate)   (exact erf gelu), f16 in / f16 out (in-place ok)
__global__ void gated_act(const _Float16* __restrict__ ff,
                          const _Float16* __restrict__ gate,
                          _Float16* __restrict__ g, size_t n) {
  size_t i = (size_t)blockIdx.x * blockDim.x + threadIdx.x;
  if (i >= n) return;
  float f  = (float)ff[i];
  float gt = (float)gate[i];
  float gl = 0.5f * f * (1.0f + erff(f * 0.70710678f));
  float sg = 1.0f / (1.0f + __expf(-gt));
  g[i] = (_Float16)(gl * sg);
}

// small dense FC: C[m,n] = act(sum_k A[m,k]W[k,n] + b[n]); lda==0 broadcasts A row
__global__ void small_fc(const float* __restrict__ A, int lda,
                         const float* __restrict__ W, int ldw,
                         const float* __restrict__ bias, float* __restrict__ C,
                         int ldc, int M, int N, int K, int act) {
  int idx = blockIdx.x * blockDim.x + threadIdx.x;
  if (idx >= M * N) return;
  int m = idx / N, n = idx % N;
  const float* a = A + (size_t)m * lda;
  float s = bias ? bias[n] : 0.0f;
  for (int k = 0; k < K; ++k) s += a[k] * W[(size_t)k * ldw + n];
  if (act == 1) s = fmaxf(s, 0.0f);
  C[(size_t)m * ldc + n] = s;
}

// attention pooling: one block per (b,h); softmax over S=512 keys
__global__ __launch_bounds__(128) void pool_attn(
    const float* __restrict__ pq, const _Float16* __restrict__ pk,
    const _Float16* __restrict__ pv, float* __restrict__ pooled,
    int S, int D, int H) {
  const int dk = D / H;
  const int b = blockIdx.x / H, h = blockIdx.x % H;
  __shared__ float sc[512];
  __shared__ float qsh[64];
  __shared__ float redA[4], redB[4];
  const int tid = threadIdx.x;
  if (tid < dk) qsh[tid] = pq[(size_t)b * D + h * dk + tid];
  __syncthreads();
  for (int s0 = tid; s0 < S; s0 += 128) {
    const _Float16* kp = pk + ((size_t)b * S + s0) * D + h * dk;
    float acc = 0.0f;
    for (int d = 0; d < dk; ++d) acc += qsh[d] * (float)kp[d];
    sc[s0] = acc * 0.125f;
  }
  __syncthreads();
  float mx = -1e30f;
  for (int s0 = tid; s0 < S; s0 += 128) mx = fmaxf(mx, sc[s0]);
  for (int off = 16; off >= 1; off >>= 1) mx = fmaxf(mx, __shfl_xor(mx, off, 32));
  int w = tid >> 5, ln = tid & 31;
  if (ln == 0) redA[w] = mx;
  __syncthreads();
  mx = fmaxf(fmaxf(redA[0], redA[1]), fmaxf(redA[2], redA[3]));
  float sm = 0.0f;
  for (int s0 = tid; s0 < S; s0 += 128) {
    float e = __expf(sc[s0] - mx);
    sc[s0] = e; sm += e;
  }
  for (int off = 16; off >= 1; off >>= 1) sm += __shfl_xor(sm, off, 32);
  if (ln == 0) redB[w] = sm;
  __syncthreads();
  sm = redB[0] + redB[1] + redB[2] + redB[3];
  float inv = 1.0f / sm;
  for (int d = tid; d < dk; d += 128) {
    float acc = 0.0f;
    for (int s0 = 0; s0 < S; ++s0)
      acc += sc[s0] * (float)pv[((size_t)b * S + s0) * D + h * dk + d];
    pooled[(size_t)b * D + h * dk + d] = acc * inv;
  }
}

// ---------------------------------------------------------------------------
extern "C" void kernel_launch(void* const* d_in, const int* in_sizes, int n_in,
                              void* d_out, int out_size, void* d_ws,
                              size_t ws_size, hipStream_t stream) {
  (void)in_sizes; (void)n_in; (void)out_size; (void)ws_size;
  const float* x       = (const float*)d_in[0];
  const float* lpe     = (const float*)d_in[1];
  const float* price_w = (const float*)d_in[2];
  const float* price_b = (const float*)d_in[3];
  const float* vol_w   = (const float*)d_in[4];
  const float* vol_b   = (const float*)d_in[5];
  const float* tech_w  = (const float*)d_in[6];
  const float* tech_b  = (const float*)d_in[7];
  const float* wq = (const float*)d_in[8];
  const float* bq = (const float*)d_in[9];
  const float* wk = (const float*)d_in[10];
  const float* bk = (const float*)d_in[11];
  const float* wv = (const float*)d_in[12];
  const float* bv = (const float*)d_in[13];
  const float* wo = (const float*)d_in[14];
  const float* bo = (const float*)d_in[15];
  const float* temp_bias = (const float*)d_in[16];
  const float* ln1_g = (const float*)d_in[17];
  const float* ln1_b = (const float*)d_in[18];
  const float* ln2_g = (const float*)d_in[19];
  const float* ln2_b = (const float*)d_in[20];
  const float* ff1_w = (const float*)d_in[21];
  const float* ff1_b = (const float*)d_in[22];
  const float* gate_w = (const float*)d_in[23];
  const float* gate_b = (const float*)d_in[24];
  const float* ff2_w = (const float*)d_in[25];
  const float* ff2_b = (const float*)d_in[26];
  const float* pool_q    = (const float*)d_in[27];
  const float* pool_in_w = (const float*)d_in[28];
  const float* pool_in_b = (const float*)d_in[29];
  const float* pool_out_w = (const float*)d_in[30];
  const float* pool_out_b = (const float*)d_in[31];
  const float* a1_w = (const float*)d_in[32];
  const float* a1_b = (const float*)d_in[33];
  const float* a2_w = (const float*)d_in[34];
  const float* a2_b = (const float*)d_in[35];
  const float* a3_w = (const float*)d_in[36];
  const float* a3_b = (const float*)d_in[37];
  const float* c1_w = (const float*)d_in[38];
  const float* c1_b = (const float*)d_in[39];
  const float* c2_w = (const float*)d_in[40];
  const float* c2_b = (const float*)d_in[41];
  const float* c3_w = (const float*)d_in[42];
  const float* c3_b = (const float*)d_in[43];
  float* out = (float*)d_out;

  constexpr int Bb = 16, Ss = 512, Ff = 2340, Dd = 512, Hh = 8, Ll = 6,
                Dff = 2048;
  const int Mrows = Bb * Ss; // 8192

  // ---- workspace layout (~105 MB peak, with reuse) ----
  char*     ws   = (char*)d_ws;
  float*    hid  = (float*)(ws + 0);                    // 16 MB f32
  _Float16* hidh = (_Float16*)(ws + ((size_t)16 << 20)); // 8 MB f16
  _Float16* qh   = (_Float16*)(ws + ((size_t)24 << 20)); // 8 MB
  _Float16* kh   = (_Float16*)(ws + ((size_t)32 << 20)); // 8 MB
  _Float16* vh   = (_Float16*)(ws + ((size_t)40 << 20)); // 8 MB
  _Float16* ctxh = (_Float16*)(ws + ((size_t)48 << 20)); // 8 MB
  _Float16* ffh  = (_Float16*)(ws + ((size_t)24 << 20)); // 32 MB (reuses q/k/v/ctx)
  _Float16* gth  = (_Float16*)(ws + ((size_t)56 << 20)); // 32 MB
  float*    tmp  = (float*)(ws + ((size_t)88 << 20));    // 16 MB f32
  float*    sm   = (float*)(ws + ((size_t)104 << 20));   // small buffers
  float* pqf     = sm;
  float* pooled  = pqf + 16 * 512;
  float* pooled2 = pooled + 16 * 512;
  float* aw1     = pooled2 + 16 * 512;
  float* aw2     = aw1 + 16 * 256;
  float* cw1     = aw2 + 16 * 128;
  float* cw2     = cw1 + 16 * 256;

  auto gg = [](int Mv, int Nv) { return dim3((Nv + 127) / 128, (Mv + 127) / 128); };

  // ---- market embedding (price | volume | technical) into tmp ----
  gemm_wmma<float><<<gg(Mrows, 128), 256, 0, stream>>>(
      x + 0, Ff, price_w, 128, price_b, tmp + 0, (_Float16*)nullptr, Dd, Mrows, 128, 5);
  gemm_wmma<float><<<gg(Mrows, 64), 256, 0, stream>>>(
      x + 5, Ff, vol_w, 64, vol_b, tmp + 128, (_Float16*)nullptr, Dd, Mrows, 64, 1);
  gemm_wmma<float><<<gg(Mrows, 320), 256, 0, stream>>>(
      x + 6, Ff, tech_w, 320, tech_b, tmp + 192, (_Float16*)nullptr, Dd, Mrows, 320, 2334);
  {
    int tot = Mrows * Dd;
    pe_add<<<(tot + 255) / 256, 256, 0, stream>>>(tmp, lpe, hid, hidh, Ss, Dd, tot);
  }

  // ---- transformer layers ----
  for (int l = 0; l < Ll; ++l) {
    const size_t wOff = (size_t)l * Dd * Dd;
    gemm_wmma<_Float16><<<gg(Mrows, Dd), 256, 0, stream>>>(
        hidh, Dd, wq + wOff, Dd, bq + l * Dd, (float*)nullptr, qh, Dd, Mrows, Dd, Dd);
    gemm_wmma<_Float16><<<gg(Mrows, Dd), 256, 0, stream>>>(
        hidh, Dd, wk + wOff, Dd, bk + l * Dd, (float*)nullptr, kh, Dd, Mrows, Dd, Dd);
    gemm_wmma<_Float16><<<gg(Mrows, Dd), 256, 0, stream>>>(
        hidh, Dd, wv + wOff, Dd, bv + l * Dd, (float*)nullptr, vh, Dd, Mrows, Dd, Dd);
    attn_wmma<<<dim3(Bb * Hh, Ss / 64), 128, 0, stream>>>(
        qh, kh, vh, ctxh, temp_bias + l * Hh, Ss, Dd, Hh);
    gemm_wmma<_Float16><<<gg(Mrows, Dd), 256, 0, stream>>>(
        ctxh, Dd, wo + wOff, Dd, bo + l * Dd, tmp, (_Float16*)nullptr, Dd, Mrows, Dd, Dd);
    add_ln<<<Mrows, 128, 0, stream>>>(tmp, hid, ln1_g + l * Dd, ln1_b + l * Dd,
                                      hid, hidh, Dd);
    const size_t fOff = (size_t)l * Dd * Dff;
    gemm_wmma<_Float16><<<gg(Mrows, Dff), 256, 0, stream>>>(
        hidh, Dd, ff1_w + fOff, Dff, ff1_b + l * Dff, (float*)nullptr, ffh, Dff,
        Mrows, Dff, Dd);
    gemm_wmma<_Float16><<<gg(Mrows, Dff), 256, 0, stream>>>(
        hidh, Dd, gate_w + fOff, Dff, gate_b + l * Dff, (float*)nullptr, gth, Dff,
        Mrows, Dff, Dd);
    {
      size_t ng = (size_t)Mrows * Dff;
      gated_act<<<(unsigned)((ng + 255) / 256), 256, 0, stream>>>(ffh, gth, ffh, ng);
    }
    gemm_wmma<_Float16><<<gg(Mrows, Dd), 256, 0, stream>>>(
        ffh, Dff, ff2_w + (size_t)l * Dff * Dd, Dd, ff2_b + l * Dd, tmp,
        (_Float16*)nullptr, Dd, Mrows, Dd, Dff);
    add_ln<<<Mrows, 128, 0, stream>>>(tmp, hid, ln2_g + l * Dd, ln2_b + l * Dd,
                                      hid, hidh, Dd);
  }

  // ---- attention pooling (pk/pv reuse q/k f16 buffers) ----
  gemm_wmma<_Float16><<<gg(Mrows, Dd), 256, 0, stream>>>(
      hidh, Dd, pool_in_w + (size_t)1 * Dd * Dd, Dd, pool_in_b + Dd,
      (float*)nullptr, qh, Dd, Mrows, Dd, Dd); // pk
  gemm_wmma<_Float16><<<gg(Mrows, Dd), 256, 0, stream>>>(
      hidh, Dd, pool_in_w + (size_t)2 * Dd * Dd, Dd, pool_in_b + 2 * Dd,
      (float*)nullptr, kh, Dd, Mrows, Dd, Dd); // pv
  small_fc<<<(16 * 512 + 127) / 128, 128, 0, stream>>>(
      pool_q, 0, pool_in_w, Dd, pool_in_b, pqf, Dd, 16, Dd, Dd, 0); // pq (bcast)
  pool_attn<<<Bb * Hh, 128, 0, stream>>>(pqf, qh, kh, pooled, Ss, Dd, Hh);
  small_fc<<<(16 * 512 + 127) / 128, 128, 0, stream>>>(
      pooled, Dd, pool_out_w, Dd, pool_out_b, pooled2, Dd, 16, Dd, Dd, 0);

  // ---- actor head -> out[:, 0:3] ----
  small_fc<<<(16 * 256 + 127) / 128, 128, 0, stream>>>(
      pooled2, Dd, a1_w, 256, a1_b, aw1, 256, 16, 256, 512, 1);
  small_fc<<<(16 * 128 + 127) / 128, 128, 0, stream>>>(
      aw1, 256, a2_w, 128, a2_b, aw2, 128, 16, 128, 256, 1);
  small_fc<<<(16 * 3 + 127) / 128, 128, 0, stream>>>(
      aw2, 128, a3_w, 3, a3_b, out, 4, 16, 3, 128, 0);
  // ---- critic head -> out[:, 3] ----
  small_fc<<<(16 * 256 + 127) / 128, 128, 0, stream>>>(
      pooled2, Dd, c1_w, 256, c1_b, cw1, 256, 16, 256, 512, 1);
  small_fc<<<(16 * 128 + 127) / 128, 128, 0, stream>>>(
      cw1, 256, c2_w, 128, c2_b, cw2, 128, 16, 128, 256, 1);
  small_fc<<<(16 * 1 + 127) / 128, 128, 0, stream>>>(
      cw2, 128, c3_w, 1, c3_b, out + 3, 4, 16, 1, 128, 0);
}
